// Mamba2_13726715478174
// MI455X (gfx1250) — matrix operand order repaired and checked by reading, not compiled, not run on hardware
//
#include <hip/hip_runtime.h>
#include <hip/hip_bf16.h>
#include <math.h>

// ---------------- problem constants (from reference) ----------------
#define DM        1024
#define D_INNER   2048
#define HEADDIM   64
#define NHEADS    32
#define DSTATE    128
#define NGROUPS   1
#define DCONV     4
#define CONV_DIM  (D_INNER + 2 * NGROUPS * DSTATE)               // 2304
#define D_IN_PROJ (2 * D_INNER + 2 * NGROUPS * DSTATE + NHEADS)  // 4384
#define NP1       4416   // D_IN_PROJ padded up to multiple of 64
#define BATCH     4
#define SEQ       1024
#define EPS_RMS   1e-5f

typedef __attribute__((ext_vector_type(16))) __bf16 v16bf;
typedef __attribute__((ext_vector_type(8)))  __bf16 v8bf;
typedef __attribute__((ext_vector_type(8)))  float  v8f;
typedef __attribute__((ext_vector_type(4)))  unsigned int v4u;
typedef __attribute__((ext_vector_type(8)))  int v8i;
typedef __attribute__((ext_vector_type(4)))  int v4i;

__device__ __forceinline__ v8f wmma_bf16(v16bf a, v16bf b, v8f c) {
  return __builtin_amdgcn_wmma_f32_16x16x32_bf16(
      /*neg_a=*/false, a, /*neg_b=*/false, b,
      /*c_mod=*/(short)0, c, /*reuse_a=*/false, /*reuse_b=*/false);
}

// =====================================================================
// bf16 conversion helpers
// =====================================================================
__global__ __launch_bounds__(256)
void cvt_to_bf16(const float* __restrict__ src, __bf16* __restrict__ dst, long n) {
  long i = (long)blockIdx.x * 256 + threadIdx.x;
  long stride = (long)gridDim.x * 256;
  for (; i < n; i += stride) dst[i] = (__bf16)src[i];
}

__global__ __launch_bounds__(256)
void cvt_pad_w_bf16(const float* __restrict__ src, __bf16* __restrict__ dst,
                    int Nsrc, int Npad, int K) {
  long n = (long)Npad * K;
  long i = (long)blockIdx.x * 256 + threadIdx.x;
  long stride = (long)gridDim.x * 256;
  for (; i < n; i += stride) {
    int r = (int)(i / K);
    dst[i] = (r < Nsrc) ? (__bf16)src[i] : (__bf16)0.f;
  }
}

// =====================================================================
// C[M,N] = A[M,K] * W[N,K]^T  (bf16 in, fp32 out)
// Block tile 128x64, 8 waves, each wave a 32x32 tile -> 4 WMMA / K-step.
// A tile: one Tensor-Data-Mover DMA per K-step (wave 0, TENSORcnt).
// B frags: plain global_load_b128 row loads straight from W.
// =====================================================================
__global__ __launch_bounds__(256)
void wmma_gemm_bf16(const __bf16* __restrict__ A, const __bf16* __restrict__ W,
                    float* __restrict__ C, int M, int N, int K) {
  __shared__ alignas(16) __bf16 lA[128 * 32];  // [m][k], row = 64B

  const int tid  = threadIdx.x;
  const int lane = tid & 31;
  const int wave = tid >> 5;
  const int wm   = wave & 3;          // M sub: wm*32
  const int wn   = wave >> 2;         // N sub: wn*32
  const int m0   = blockIdx.x * 128;
  const int n0   = blockIdx.y * 64;
  const int half = lane >> 4;
  const int l16  = lane & 15;

  v8f acc00 = {0.f,0.f,0.f,0.f,0.f,0.f,0.f,0.f};
  v8f acc01 = acc00, acc10 = acc00, acc11 = acc00;

  // ---- TDM descriptor group 1 (constant over K loop): ISA 8.4 ----
  // data_size=2B; tile 32(k) x 128(rows); tensor_dim0=32, tensor_dim1=128;
  // tensor_dim0_stride = K elements.
  v8i g1;
  g1[0] = (int)(1u << 16);                    // data_size=1 (2 bytes)
  g1[1] = (int)(32u << 16);                   // tensor_dim0[15:0] -> bits[31:16]
  g1[2] = (int)(128u << 16);                  // dim0 hi=0 | tensor_dim1 lo=128
  g1[3] = (int)(32u << 16);                   // dim1 hi=0 | tile_dim0=32
  g1[4] = (int)(128u);                        // tile_dim1=128, tile_dim2=0
  g1[5] = (int)(unsigned)K;                   // tensor_dim0_stride lo (elements)
  g1[6] = 0;                                  // stride0 hi | stride1 lo
  g1[7] = 0;
  v4i g2 = {0, 0, 0, 0};                      // dims 2/3 unused (tile_dim3=0)
  v4i g3 = {0, 0, 0, 0};                      // dim 4 unused

  for (int k0 = 0; k0 < K; k0 += 32) {
    if (wave == 0) {
      // ---- TDM group 0: count=1, lds_addr, 57-bit global addr, type=2 ----
      unsigned long long ga =
          (unsigned long long)(const void*)(A + (size_t)m0 * K + k0);
      v4u g0;
      g0[0] = 1u;
      g0[1] = (unsigned int)(unsigned long long)(void*)lA;   // LDS offset
      g0[2] = (unsigned int)ga;
      g0[3] = (unsigned int)((ga >> 32) & 0x1ffffffu) | (2u << 30);
      asm volatile("tensor_load_to_lds %0, %1, %2, %3"
                   :: "s"(g0), "s"(g1), "s"(g2), "s"(g3) : "memory");
      __builtin_amdgcn_s_wait_tensorcnt(0);
    }
    __syncthreads();   // lA now holds this K-step's A tile for all waves

    // ---- B fragments: plain 16B row loads from W[n][k], two K-halves each ----
    union { v16bf v; uint4 u[2]; } b0f, b1f;
    {
      const char* r0 = (const char*)(W + (size_t)(n0 + wn * 32 + l16) * K + k0);
      const char* r1 = (const char*)(W + (size_t)(n0 + wn * 32 + 16 + l16) * K + k0);
      const char* p00 = r0 + half * 16;               // N 0..15,  K 0..15
      const char* p01 = r0 + 32 + half * 16;          // N 0..15,  K 16..31
      const char* p10 = r1 + half * 16;               // N 16..31, K 0..15
      const char* p11 = r1 + 32 + half * 16;          // N 16..31, K 16..31
      if (k0 + 32 < K) __builtin_prefetch(r0 + 64, 0, 0);
      asm volatile("global_load_b128 %0, %1, off" : "=v"(b0f.u[0]) : "v"(p00));
      asm volatile("global_load_b128 %0, %1, off" : "=v"(b0f.u[1]) : "v"(p01));
      asm volatile("global_load_b128 %0, %1, off" : "=v"(b1f.u[0]) : "v"(p10));
      asm volatile("global_load_b128 %0, %1, off" : "=v"(b1f.u[1]) : "v"(p11));
    }
    // Inline-asm loads are not counter-tracked by the compiler. Volatile asm
    // blocks stay in program order, so this wait is after all B loads; the
    // "memory" clobber pins the (compiler-tracked) A-fragment LDS loads below
    // it, and the WMMAs data-depend on those -> WMMA executes after the wait.
    asm volatile("s_wait_loadcnt 0x0" ::: "memory");

    // ---- A fragments from LDS (ISA 7.12.2 16-bit layout, 16B reads) ----
    union { v16bf v; uint4 u[2]; } a0f, a1f;
    const char* pa0 = (const char*)lA + (wm * 32 + l16) * 64;
    a0f.u[0] = *(const uint4*)(pa0 + half * 16);
    a0f.u[1] = *(const uint4*)(pa0 + 32 + half * 16);
    const char* pa1 = pa0 + 16 * 64;
    a1f.u[0] = *(const uint4*)(pa1 + half * 16);
    a1f.u[1] = *(const uint4*)(pa1 + 32 + half * 16);

    acc00 = wmma_bf16(a0f.v, b0f.v, acc00);
    acc01 = wmma_bf16(a0f.v, b1f.v, acc01);
    acc10 = wmma_bf16(a1f.v, b0f.v, acc10);
    acc11 = wmma_bf16(a1f.v, b1f.v, acc11);
    __syncthreads();   // all reads of lA done before next TDM overwrites it
  }

  // ---- epilogue: lanes 0-15 -> M=r, lanes 16-31 -> M=8+r; N=l16 ----
  {
    const int mb = m0 + wm * 32;
    const int nb = n0 + wn * 32;
    #pragma unroll
    for (int r = 0; r < 8; ++r) {
      int mr = (half ? (r + 8) : r);
      C[(size_t)(mb + mr)      * N + nb + l16]      = acc00[r];
      C[(size_t)(mb + mr)      * N + nb + 16 + l16] = acc01[r];
      C[(size_t)(mb + 16 + mr) * N + nb + l16]      = acc10[r];
      C[(size_t)(mb + 16 + mr) * N + nb + 16 + l16] = acc11[r];
    }
  }
}

// =====================================================================
// depthwise causal conv (4 taps) + bias + SiLU, plus dt = softplus(...)
// =====================================================================
__global__ __launch_bounds__(256)
void conv_silu_dt(const float* __restrict__ zxbcdt,
                  const float* __restrict__ conv_w,
                  const float* __restrict__ conv_b,
                  const float* __restrict__ dt_bias,
                  float* __restrict__ xBCconv,
                  float* __restrict__ dtbuf) {
  const int bl = blockIdx.x;
  const int b  = bl / SEQ;
  const int l  = bl % SEQ;
  float* orow = xBCconv + (size_t)bl * CONV_DIM;

  for (int c = threadIdx.x; c < CONV_DIM; c += 256) {
    float acc = conv_b[c];
    #pragma unroll
    for (int j = 0; j < DCONV; ++j) {
      int ll = l - (DCONV - 1) + j;
      if (ll >= 0) {
        float xv = zxbcdt[((size_t)b * SEQ + ll) * NP1 + D_INNER + c];
        acc = fmaf(conv_w[c * DCONV + j], xv, acc);
      }
    }
    orow[c] = acc / (1.f + __expf(-acc));   // SiLU
  }
  for (int hh = threadIdx.x; hh < NHEADS; hh += 256) {
    float v = zxbcdt[(size_t)bl * NP1 + D_INNER + CONV_DIM + hh] + dt_bias[hh];
    dtbuf[(size_t)bl * NHEADS + hh] = (v > 20.f) ? v : log1pf(__expf(v));
  }
}

// =====================================================================
// sequential SSM scan: one block per (batch, head); state in registers
// =====================================================================
__global__ __launch_bounds__(256)
void ssm_scan(const float* __restrict__ xBCconv,
              const float* __restrict__ dtbuf,
              const float* __restrict__ A_log,
              const float* __restrict__ Dvec,
              float* __restrict__ yD) {
  const int bh = blockIdx.x;
  const int b  = bh >> 5;
  const int hh = bh & 31;

  __shared__ float sB[DSTATE];
  __shared__ float sC[DSTATE];
  __shared__ float sPart[HEADDIM * 4];

  const int tid = threadIdx.x;
  const int p   = tid >> 2;
  const int q   = tid & 3;
  const int nb  = q * 32;

  float h[32];
  #pragma unroll
  for (int j = 0; j < 32; ++j) h[j] = 0.f;

  const float Acoef = -__expf(A_log[hh]);
  const float Dh    = Dvec[hh];

  for (int l = 0; l < SEQ; ++l) {
    const float* row = xBCconv + ((size_t)b * SEQ + l) * CONV_DIM;
    if (tid < DSTATE)            sB[tid] = row[D_INNER + tid];
    else if (tid < 2 * DSTATE)   sC[tid - DSTATE] = row[D_INNER + DSTATE + (tid - DSTATE)];

    const float dtv = dtbuf[((size_t)b * SEQ + l) * NHEADS + hh];
    const float dA  = __expf(dtv * Acoef);
    const float xv  = row[hh * HEADDIM + p];
    const float dtx = dtv * xv;
    __syncthreads();

    float accl = 0.f;
    #pragma unroll
    for (int j = 0; j < 32; ++j) {
      h[j] = fmaf(dA, h[j], dtx * sB[nb + j]);
      accl = fmaf(h[j], sC[nb + j], accl);
    }
    sPart[p * 4 + q] = accl;
    __syncthreads();

    if (q == 0) {
      float yv = sPart[p * 4] + sPart[p * 4 + 1] + sPart[p * 4 + 2] + sPart[p * 4 + 3];
      yv = fmaf(Dh, xv, yv);
      yD[((size_t)b * SEQ + l) * D_INNER + hh * HEADDIM + p] = yv;
    }
    __syncthreads();
  }
}

// =====================================================================
// yz = RMSNorm( y * silu(z) ) * norm_w  -> written directly as bf16
// =====================================================================
__global__ __launch_bounds__(256)
void gate_rmsnorm(const float* __restrict__ yD,
                  const float* __restrict__ zxbcdt,
                  const float* __restrict__ norm_w,
                  __bf16* __restrict__ yz_bf) {
  const int r = blockIdx.x;
  __shared__ float red[256];
  const float* yrow = yD + (size_t)r * D_INNER;
  const float* zrow = zxbcdt + (size_t)r * NP1;

  float loc[8];
  float ss = 0.f;
  #pragma unroll
  for (int i = 0; i < 8; ++i) {
    int idx = threadIdx.x + i * 256;
    float z = zrow[idx];
    float v = yrow[idx] * (z / (1.f + __expf(-z)));
    loc[i] = v;
    ss = fmaf(v, v, ss);
  }
  red[threadIdx.x] = ss;
  __syncthreads();
  #pragma unroll
  for (int s = 128; s > 0; s >>= 1) {
    if (threadIdx.x < s) red[threadIdx.x] += red[threadIdx.x + s];
    __syncthreads();
  }
  const float rms = rsqrtf(red[0] * (1.f / (float)D_INNER) + EPS_RMS);
  #pragma unroll
  for (int i = 0; i < 8; ++i) {
    int idx = threadIdx.x + i * 256;
    yz_bf[(size_t)r * D_INNER + idx] = (__bf16)(loc[i] * rms * norm_w[idx]);
  }
}

// =====================================================================
// Host-side launch
// =====================================================================
extern "C" void kernel_launch(void* const* d_in, const int* in_sizes, int n_in,
                              void* d_out, int out_size, void* d_ws, size_t ws_size,
                              hipStream_t stream) {
  const float* x         = (const float*)d_in[0];
  const float* in_proj_w = (const float*)d_in[1];
  const float* conv_w    = (const float*)d_in[2];
  const float* conv_b    = (const float*)d_in[3];
  const float* dt_bias   = (const float*)d_in[4];
  const float* A_log     = (const float*)d_in[5];
  const float* Dvec      = (const float*)d_in[6];
  const float* norm_w    = (const float*)d_in[7];
  const float* out_proj_w= (const float*)d_in[8];
  float* out             = (float*)d_out;

  const int M = BATCH * SEQ;  // 4096

  // ---- workspace layout ----
  char* ws = (char*)d_ws;
  float*  zxbcdt  = (float*)ws;                 ws += (size_t)M * NP1 * 4;
  float*  xBCconv = (float*)ws;                 ws += (size_t)M * CONV_DIM * 4;
  float*  dtbuf   = (float*)ws;                 ws += (size_t)M * NHEADS * 4;
  float*  yD      = (float*)ws;                 ws += (size_t)M * D_INNER * 4;
  __bf16* xbf     = (__bf16*)ws;                ws += (size_t)M * DM * 2;
  __bf16* w1bf    = (__bf16*)ws;                ws += (size_t)NP1 * DM * 2;
  __bf16* w2bf    = (__bf16*)ws;                ws += (size_t)DM * D_INNER * 2;
  __bf16* yzbf    = (__bf16*)ws;                ws += (size_t)M * D_INNER * 2;

  // 0) fp32 -> bf16 conversions
  cvt_to_bf16 <<<2048, 256, 0, stream>>>(x, xbf, (long)M * DM);
  cvt_pad_w_bf16<<<2048, 256, 0, stream>>>(in_proj_w, w1bf, D_IN_PROJ, NP1, DM);
  cvt_to_bf16 <<<2048, 256, 0, stream>>>(out_proj_w, w2bf, (long)DM * D_INNER);

  // 1) in_proj GEMM
  {
    dim3 grid(M / 128, NP1 / 64);   // 32 x 69
    wmma_gemm_bf16<<<grid, 256, 0, stream>>>(xbf, w1bf, zxbcdt, M, NP1, DM);
  }
  // 2) conv + SiLU + softplus(dt)
  conv_silu_dt<<<M, 256, 0, stream>>>(zxbcdt, conv_w, conv_b, dt_bias, xBCconv, dtbuf);
  // 3) SSM recurrence
  ssm_scan<<<BATCH * NHEADS, 256, 0, stream>>>(xBCconv, dtbuf, A_log, Dvec, yD);
  // 4) gate + RMSNorm -> bf16
  gate_rmsnorm<<<M, 256, 0, stream>>>(yD, zxbcdt, norm_w, yzbf);
  // 5) out_proj GEMM
  {
    dim3 grid(M / 128, DM / 64);    // 32 x 16
    wmma_gemm_bf16<<<grid, 256, 0, stream>>>(yzbf, w2bf, out, M, DM, D_INNER);
  }
}
